// MPNNLSTM_3109556322622
// MI455X (gfx1250) — compile-verified, matrix operand order checked by reading, not run on hardware
//
#include <hip/hip_runtime.h>
#include <math.h>

// ---------------------------------------------------------------------------
// Types for CDNA5 WMMA (wave32)
// ---------------------------------------------------------------------------
typedef __bf16 bf16;
typedef __bf16 v16bf __attribute__((ext_vector_type(16)));
typedef float  v8f   __attribute__((ext_vector_type(8)));

#define NUM_NODES 10000
#define WINDOW    8
#define IN_C      8
#define HID       64
#define N_TOT     (WINDOW * NUM_NODES)       // 80000
#define N_EDGES   1280000
#define GATES     (4 * HID)                  // 256
#define OUT_COLS  (2 * HID + IN_C + WINDOW - 1)  // 143
#define EPS_BN    1e-5f

static __device__ __forceinline__ float sigmoidf_(float x) {
    return 1.0f / (1.0f + __expf(-x));
}

// ---------------------------------------------------------------------------
// Utility kernels
// ---------------------------------------------------------------------------
__global__ void k_zero_u32(unsigned int* __restrict__ p, long long n) {
    long long i = (long long)blockIdx.x * blockDim.x + threadIdx.x;
    if (i < n) p[i] = 0u;
}

// wt[n*K + k] = w[k*N + n]   (transpose + convert, for w2 [K,N] -> [N,K])
__global__ void k_f32_to_bf16_T(const float* __restrict__ w, bf16* __restrict__ wt, int K, int N) {
    int i = blockIdx.x * blockDim.x + threadIdx.x;
    if (i < K * N) {
        int n = i / K, k = i % K;
        wt[i] = (bf16)w[k * N + n];
    }
}

// Pack [N, K1+K2] bf16: out[n][k] = k<K1 ? a[n][k] : b[n][k-K1]
// (a = wih [N,K1], b = whh [N,K2], both already [N,K] row-major)
__global__ void k_pack_bf16(const float* __restrict__ a, const float* __restrict__ b,
                            bf16* __restrict__ outp, int N, int K1, int K2) {
    int KT = K1 + K2;
    int i = blockIdx.x * blockDim.x + threadIdx.x;
    if (i >= N * KT) return;
    int n = i / KT, k = i % KT;
    float v = (k < K1) ? a[n * K1 + k] : b[n * K2 + (k - K1)];
    outp[i] = (bf16)v;
}

// ---------------------------------------------------------------------------
// Degree / normalization
// ---------------------------------------------------------------------------
__global__ void k_deg_accum(const int* __restrict__ dst, const float* __restrict__ ew,
                            float* __restrict__ deg, int E) {
    int e = blockIdx.x * blockDim.x + threadIdx.x;
    if (e < E) atomicAdd(&deg[dst[e]], ew[e]);
}

__global__ void k_deg_to_dinv(float* __restrict__ deg, int n) {
    int i = blockIdx.x * blockDim.x + threadIdx.x;
    if (i < n) deg[i] = rsqrtf(deg[i] + 1.0f);
}

// ---------------------------------------------------------------------------
// GCN layer 1 transform: h = X[80000,8] @ w1[8,64]  (K=8, memory bound: scalar)
// ---------------------------------------------------------------------------
__global__ void k_gemm_in8(const float* __restrict__ X, const float* __restrict__ w,
                           float* __restrict__ hout, int M) {
    int i = blockIdx.x * blockDim.x + threadIdx.x;
    if (i >= M * HID) return;
    int r = i / HID, c = i % HID;
    const float* xr = X + (long long)r * IN_C;
    float acc = 0.0f;
#pragma unroll
    for (int k = 0; k < IN_C; ++k) acc = fmaf(xr[k], w[k * HID + c], acc);
    hout[i] = acc;
}

// ---------------------------------------------------------------------------
// Edge aggregation: agg[dst] += dinv[src]*ew*dinv[dst] * h[src]
// One thread handles 4 channels of one edge (float4 gather + 4 atomics).
// ---------------------------------------------------------------------------
__global__ void k_edge_agg(const int* __restrict__ src, const int* __restrict__ dst,
                           const float* __restrict__ ew, const float* __restrict__ dinv,
                           const float* __restrict__ h, float* __restrict__ agg, int E) {
    long long tid = (long long)blockIdx.x * blockDim.x + threadIdx.x;
    long long e = tid >> 4;
    if (e >= E) return;
    int part = (int)(tid & 15);
    int s = src[e], d = dst[e];
    float coef = dinv[s] * ew[e] * dinv[d];
    const float4 v = *(const float4*)(h + (long long)s * HID + part * 4);
    float* ap = agg + (long long)d * HID + part * 4;
    atomicAdd(ap + 0, coef * v.x);
    atomicAdd(ap + 1, coef * v.y);
    atomicAdd(ap + 2, coef * v.z);
    atomicAdd(ap + 3, coef * v.w);
}

// out = relu(agg + h*dinv^2 + b), in-place in agg
__global__ void k_combine_relu(float* __restrict__ agg, const float* __restrict__ h,
                               const float* __restrict__ dinv, const float* __restrict__ b, int M) {
    long long i = (long long)blockIdx.x * blockDim.x + threadIdx.x;
    if (i >= (long long)M * HID) return;
    int r = (int)(i / HID), c = (int)(i % HID);
    float di = dinv[r];
    float v = agg[i] + h[i] * di * di + b[c];
    agg[i] = v > 0.0f ? v : 0.0f;
}

// ---------------------------------------------------------------------------
// BatchNorm: one block per channel, biased variance over M rows
// ---------------------------------------------------------------------------
__global__ void k_bn_stats(const float* __restrict__ x, float* __restrict__ mu,
                           float* __restrict__ rstd, int M) {
    __shared__ float ssum[256];
    __shared__ float ssq[256];
    int c = blockIdx.x;
    float s = 0.0f, q = 0.0f;
    for (int r = threadIdx.x; r < M; r += blockDim.x) {
        float v = x[(long long)r * HID + c];
        s += v; q += v * v;
    }
    ssum[threadIdx.x] = s; ssq[threadIdx.x] = q;
    __syncthreads();
    for (int st = 128; st > 0; st >>= 1) {
        if (threadIdx.x < st) {
            ssum[threadIdx.x] += ssum[threadIdx.x + st];
            ssq[threadIdx.x]  += ssq[threadIdx.x + st];
        }
        __syncthreads();
    }
    if (threadIdx.x == 0) {
        float m = ssum[0] / (float)M;
        float var = ssq[0] / (float)M - m * m;
        mu[c] = m;
        rstd[c] = rsqrtf(var + EPS_BN);
    }
}

// y = (x-mu)*rstd*g + be, written as bf16 into xc[:, colOff:colOff+64] (ld=128)
__global__ void k_bn_apply_bf16(const float* __restrict__ x, const float* __restrict__ mu,
                                const float* __restrict__ rstd, const float* __restrict__ g,
                                const float* __restrict__ be, bf16* __restrict__ xc,
                                int colOff, int M) {
    long long i = (long long)blockIdx.x * blockDim.x + threadIdx.x;
    if (i >= (long long)M * HID) return;
    int r = (int)(i / HID), c = (int)(i % HID);
    float y = (x[i] - mu[c]) * rstd[c] * g[c] + be[c];
    xc[(long long)r * (2 * HID) + colOff + c] = (bf16)y;
}

// ---------------------------------------------------------------------------
// Dual-segment WMMA bf16 GEMM:
//   C[M,N] = A1[M,K1] @ Bt[:, 0:K1]^T  +  A2[M,K2] @ Bt[:, K1:K1+K2]^T
// Bt stored [N, K1+K2] row-major (ldb = K1+K2). A2/K2 may be null/0.
// One wave computes a 16x64 strip of C: A fragment loaded once per K-step and
// reused across 4 independent v_wmma_f32_16x16x32_bf16 accumulator chains.
// Requirements: M%16==0, N%64==0, K1%32==0, K2%32==0.
//
// CDNA5 wave32 fragment layout (ISA 7.12.2):
//   A: lane (m = lane&15, half = lane>>4); elems 0..7 -> K=half*8+e,
//      elems 8..15 -> K=16+half*8+(e-8)   (two contiguous 16B runs)
//   B mirrors A with N in place of M (Bt stored [N,K] row-major)
//   C/D: lane holds col n=lane&15; elem i -> row (half*8 + i)
// ---------------------------------------------------------------------------
static __device__ __forceinline__ v16bf load_frag_bf16(const bf16* __restrict__ base,
                                                       int ld, int m, int half) {
    const bf16* p = base + (long long)m * ld + half * 8;
    v16bf r;
#pragma unroll
    for (int e = 0; e < 8; ++e) r[e] = p[e];
#pragma unroll
    for (int e = 0; e < 8; ++e) r[8 + e] = p[16 + e];
    return r;
}

__global__ void k_gemm_bf16_dual(const bf16* __restrict__ A1, int lda1, int K1,
                                 const bf16* __restrict__ A2, int lda2, int K2,
                                 const bf16* __restrict__ Bt, int ldb,
                                 float* __restrict__ C, int ldc, int M, int N) {
    int wave = threadIdx.x >> 5;
    int lane = threadIdx.x & 31;
    int stripsN = N >> 6;                       // 16x64 strips along N
    int total = (M >> 4) * stripsN;
    int strip = blockIdx.x * (blockDim.x >> 5) + wave;
    if (strip >= total) return;                 // wave-uniform: EXEC stays all-ones
    int tm = strip / stripsN;
    int tn = strip - tm * stripsN;              // strip col index (64-wide)
    int m0 = lane & 15;
    int half = lane >> 4;

    const bf16* B0 = Bt + (long long)(tn * 64) * ldb;    // 64 consecutive B rows

    v8f acc0 = {0.f, 0.f, 0.f, 0.f, 0.f, 0.f, 0.f, 0.f};
    v8f acc1 = acc0, acc2 = acc0, acc3 = acc0;

    // segment 1
    {
        const bf16* Abase = A1 + (long long)(tm * 16) * lda1;
        for (int k0 = 0; k0 < K1; k0 += 32) {
            v16bf a  = load_frag_bf16(Abase + k0, lda1, m0, half);
            v16bf b0 = load_frag_bf16(B0 + k0,                       ldb, m0, half);
            v16bf b1 = load_frag_bf16(B0 + (long long)16 * ldb + k0, ldb, m0, half);
            v16bf b2 = load_frag_bf16(B0 + (long long)32 * ldb + k0, ldb, m0, half);
            v16bf b3 = load_frag_bf16(B0 + (long long)48 * ldb + k0, ldb, m0, half);
            acc0 = __builtin_amdgcn_wmma_f32_16x16x32_bf16(false, a, false, b0, (short)0, acc0, false, false);
            acc1 = __builtin_amdgcn_wmma_f32_16x16x32_bf16(false, a, false, b1, (short)0, acc1, false, false);
            acc2 = __builtin_amdgcn_wmma_f32_16x16x32_bf16(false, a, false, b2, (short)0, acc2, false, false);
            acc3 = __builtin_amdgcn_wmma_f32_16x16x32_bf16(false, a, false, b3, (short)0, acc3, false, false);
        }
    }
    // segment 2 (recurrent input; skipped when K2 == 0)
    {
        const bf16* Abase = A2 + (long long)(tm * 16) * lda2;
        for (int k0 = 0; k0 < K2; k0 += 32) {
            v16bf a  = load_frag_bf16(Abase + k0, lda2, m0, half);
            v16bf b0 = load_frag_bf16(B0 + K1 + k0,                       ldb, m0, half);
            v16bf b1 = load_frag_bf16(B0 + (long long)16 * ldb + K1 + k0, ldb, m0, half);
            v16bf b2 = load_frag_bf16(B0 + (long long)32 * ldb + K1 + k0, ldb, m0, half);
            v16bf b3 = load_frag_bf16(B0 + (long long)48 * ldb + K1 + k0, ldb, m0, half);
            acc0 = __builtin_amdgcn_wmma_f32_16x16x32_bf16(false, a, false, b0, (short)0, acc0, false, false);
            acc1 = __builtin_amdgcn_wmma_f32_16x16x32_bf16(false, a, false, b1, (short)0, acc1, false, false);
            acc2 = __builtin_amdgcn_wmma_f32_16x16x32_bf16(false, a, false, b2, (short)0, acc2, false, false);
            acc3 = __builtin_amdgcn_wmma_f32_16x16x32_bf16(false, a, false, b3, (short)0, acc3, false, false);
        }
    }

    float* Cp = C + (long long)(tm * 16) * ldc + tn * 64 + m0;
#pragma unroll
    for (int i = 0; i < 8; ++i) {
        float* row = Cp + (long long)(half * 8 + i) * ldc;
        row[0]  = acc0[i];
        row[16] = acc1[i];
        row[32] = acc2[i];
        row[48] = acc3[i];
    }
}

// ---------------------------------------------------------------------------
// LSTM gate nonlinearity (torch gate order i,f,g,o over the 4*HID gate dim)
// ---------------------------------------------------------------------------
__global__ void k_lstm_gate(const float* __restrict__ z, const float* __restrict__ bih,
                            const float* __restrict__ bhh, float* __restrict__ c,
                            bf16* __restrict__ hbf, bf16* __restrict__ ys,
                            float* __restrict__ hout, int houtOff) {
    int idx = blockIdx.x * blockDim.x + threadIdx.x;
    if (idx >= NUM_NODES * HID) return;
    int n = idx / HID, j = idx % HID;
    const float* zr = z + (long long)n * GATES;
    float zi = zr[j]           + bih[j]           + bhh[j];
    float zf = zr[HID + j]     + bih[HID + j]     + bhh[HID + j];
    float zg = zr[2 * HID + j] + bih[2 * HID + j] + bhh[2 * HID + j];
    float zo = zr[3 * HID + j] + bih[3 * HID + j] + bhh[3 * HID + j];
    float cn = sigmoidf_(zf) * c[idx] + sigmoidf_(zi) * tanhf(zg);
    float hn = sigmoidf_(zo) * tanhf(cn);
    c[idx] = cn;
    hbf[idx] = (bf16)hn;
    if (ys)   ys[idx] = (bf16)hn;
    if (hout) hout[(long long)n * OUT_COLS + houtOff + j] = hn;
}

// ---------------------------------------------------------------------------
// Skip features: out[n,128:136] = X[n,0:8]; out[n,136+t-1] = X[t*10000+n,7]
// ---------------------------------------------------------------------------
__global__ void k_sfeat(const float* __restrict__ X, float* __restrict__ out) {
    int n = blockIdx.x * blockDim.x + threadIdx.x;
    if (n >= NUM_NODES) return;
    float* o = out + (long long)n * OUT_COLS + 2 * HID;
#pragma unroll
    for (int ci = 0; ci < IN_C; ++ci) o[ci] = X[(long long)n * IN_C + ci];
#pragma unroll
    for (int t = 1; t < WINDOW; ++t)
        o[IN_C + t - 1] = X[((long long)t * NUM_NODES + n) * IN_C + (IN_C - 1)];
}

// ---------------------------------------------------------------------------
// Host orchestration
// ---------------------------------------------------------------------------
static inline dim3 nblk(long long n, int t) { return dim3((unsigned)((n + t - 1) / t)); }

extern "C" void kernel_launch(void* const* d_in, const int* in_sizes, int n_in,
                              void* d_out, int out_size, void* d_ws, size_t ws_size,
                              hipStream_t stream) {
    const float* X    = (const float*)d_in[0];
    const int*   ei   = (const int*)d_in[1];
    const float* ew   = (const float*)d_in[2];
    const float* w1   = (const float*)d_in[3];
    const float* b1   = (const float*)d_in[4];
    const float* w2   = (const float*)d_in[5];
    const float* b2   = (const float*)d_in[6];
    const float* g1   = (const float*)d_in[7];
    const float* be1  = (const float*)d_in[8];
    const float* g2   = (const float*)d_in[9];
    const float* be2  = (const float*)d_in[10];
    const float* wih1 = (const float*)d_in[11];
    const float* whh1 = (const float*)d_in[12];
    const float* bih1 = (const float*)d_in[13];
    const float* bhh1 = (const float*)d_in[14];
    const float* wih2 = (const float*)d_in[15];
    const float* whh2 = (const float*)d_in[16];
    const float* bih2 = (const float*)d_in[17];
    const float* bhh2 = (const float*)d_in[18];
    float* out = (float*)d_out;
    const int* src  = ei;
    const int* dstv = ei + N_EDGES;

    // ---- workspace carve-out
    char* ws = (char*)d_ws;
    size_t off = 0;
    auto carve = [&](size_t bytes) -> void* {
        void* p = ws + off;
        off = (off + bytes + 255) & ~(size_t)255;
        return p;
    };
    float* deg    = (float*)carve((size_t)N_TOT * 4);                 // -> dinv in place
    float* bufH   = (float*)carve((size_t)N_TOT * HID * 4);
    float* bufAgg = (float*)carve((size_t)N_TOT * HID * 4);
    bf16*  xc     = (bf16*) carve((size_t)N_TOT * 2 * HID * 2);       // [80000,128] bf16
    bf16*  ys     = (bf16*) carve((size_t)N_TOT * HID * 2);           // LSTM1 outputs, all t
    float* zbuf   = (float*)carve((size_t)NUM_NODES * GATES * 4);
    float* cbuf   = (float*)carve((size_t)NUM_NODES * HID * 4);
    bf16*  hbf    = (bf16*) carve((size_t)NUM_NODES * HID * 2);
    bf16*  w2t    = (bf16*) carve((size_t)HID * HID * 2);
    bf16*  wcat1  = (bf16*) carve((size_t)GATES * (2 * HID + HID) * 2);  // [256,192]
    bf16*  wcat2  = (bf16*) carve((size_t)GATES * (HID + HID) * 2);      // [256,128]
    float* mu     = (float*)carve(HID * 4);
    float* rstd   = (float*)carve(HID * 4);

    const dim3 B256(256), B128(128);
    const long long NH = (long long)N_TOT * HID;

    // ---- degrees -> dinv
    k_zero_u32<<<nblk(N_TOT, 256), B256, 0, stream>>>((unsigned*)deg, N_TOT);
    k_deg_accum<<<nblk(N_EDGES, 256), B256, 0, stream>>>(dstv, ew, deg, N_EDGES);
    k_deg_to_dinv<<<nblk(N_TOT, 256), B256, 0, stream>>>(deg, N_TOT);
    const float* dinv = deg;

    // ---- weight conversions / packing (bf16)
    k_f32_to_bf16_T<<<nblk(HID * HID, 256), B256, 0, stream>>>(w2, w2t, HID, HID);
    k_pack_bf16<<<nblk(GATES * (2 * HID + HID), 256), B256, 0, stream>>>(wih1, whh1, wcat1, GATES, 2 * HID, HID);
    k_pack_bf16<<<nblk(GATES * (HID + HID), 256), B256, 0, stream>>>(wih2, whh2, wcat2, GATES, HID, HID);

    // ================= GCN layer 1 =================
    k_gemm_in8<<<nblk(NH, 256), B256, 0, stream>>>(X, w1, bufH, N_TOT);
    k_zero_u32<<<nblk(NH, 256), B256, 0, stream>>>((unsigned*)bufAgg, NH);
    k_edge_agg<<<nblk((long long)N_EDGES * 16, 256), B256, 0, stream>>>(src, dstv, ew, dinv, bufH, bufAgg, N_EDGES);
    k_combine_relu<<<nblk(NH, 256), B256, 0, stream>>>(bufAgg, bufH, dinv, b1, N_TOT);
    k_bn_stats<<<dim3(HID), B256, 0, stream>>>(bufAgg, mu, rstd, N_TOT);
    k_bn_apply_bf16<<<nblk(NH, 256), B256, 0, stream>>>(bufAgg, mu, rstd, g1, be1, xc, 0, N_TOT);

    // ================= GCN layer 2 =================
    {
        int strips = (N_TOT / 16) * (HID / 64);   // 5000 strips, 4 waves/block
        k_gemm_bf16_dual<<<nblk(strips, 4), B128, 0, stream>>>(
            xc, 2 * HID, HID, nullptr, 0, 0, w2t, HID, bufH, HID, N_TOT, HID);
    }
    k_zero_u32<<<nblk(NH, 256), B256, 0, stream>>>((unsigned*)bufAgg, NH);
    k_edge_agg<<<nblk((long long)N_EDGES * 16, 256), B256, 0, stream>>>(src, dstv, ew, dinv, bufH, bufAgg, N_EDGES);
    k_combine_relu<<<nblk(NH, 256), B256, 0, stream>>>(bufAgg, bufH, dinv, b2, N_TOT);
    k_bn_stats<<<dim3(HID), B256, 0, stream>>>(bufAgg, mu, rstd, N_TOT);
    k_bn_apply_bf16<<<nblk(NH, 256), B256, 0, stream>>>(bufAgg, mu, rstd, g2, be2, xc, HID, N_TOT);

    // ================= LSTM layer 1 =================
    k_zero_u32<<<nblk((long long)NUM_NODES * HID / 2, 256), B256, 0, stream>>>((unsigned*)hbf, (long long)NUM_NODES * HID / 2);
    k_zero_u32<<<nblk((long long)NUM_NODES * HID, 256), B256, 0, stream>>>((unsigned*)cbuf, (long long)NUM_NODES * HID);
    {
        int strips = (NUM_NODES / 16) * (GATES / 64);  // 625*4 = 2500 strips
        for (int t = 0; t < WINDOW; ++t) {
            const bf16* xt = xc + (size_t)t * NUM_NODES * 2 * HID;
            k_gemm_bf16_dual<<<nblk(strips, 4), B128, 0, stream>>>(
                xt, 2 * HID, 2 * HID, hbf, HID, HID,
                wcat1, 2 * HID + HID, zbuf, GATES, NUM_NODES, GATES);
            k_lstm_gate<<<nblk((long long)NUM_NODES * HID, 256), B256, 0, stream>>>(
                zbuf, bih1, bhh1, cbuf, hbf,
                ys + (size_t)t * NUM_NODES * HID,
                (t == WINDOW - 1) ? out : nullptr, 0);
        }
    }

    // ================= LSTM layer 2 =================
    k_zero_u32<<<nblk((long long)NUM_NODES * HID / 2, 256), B256, 0, stream>>>((unsigned*)hbf, (long long)NUM_NODES * HID / 2);
    k_zero_u32<<<nblk((long long)NUM_NODES * HID, 256), B256, 0, stream>>>((unsigned*)cbuf, (long long)NUM_NODES * HID);
    {
        int strips = (NUM_NODES / 16) * (GATES / 64);
        for (int t = 0; t < WINDOW; ++t) {
            const bf16* xt = ys + (size_t)t * NUM_NODES * HID;
            k_gemm_bf16_dual<<<nblk(strips, 4), B128, 0, stream>>>(
                xt, HID, HID, hbf, HID, HID,
                wcat2, HID + HID, zbuf, GATES, NUM_NODES, GATES);
            k_lstm_gate<<<nblk((long long)NUM_NODES * HID, 256), B256, 0, stream>>>(
                zbuf, bih2, bhh2, cbuf, hbf, nullptr,
                (t == WINDOW - 1) ? out : nullptr, HID);
        }
    }

    // ---- skip features
    k_sfeat<<<nblk(NUM_NODES, 256), B256, 0, stream>>>(X, out);
}
